// GAT_63608465654252
// MI455X (gfx1250) — matrix operand order repaired and checked by reading, not compiled, not run on hardware
//
#include <hip/hip_runtime.h>

typedef __attribute__((ext_vector_type(16))) _Float16 v16h;
typedef __attribute__((ext_vector_type(8)))  _Float16 v8h;
typedef __attribute__((ext_vector_type(8)))  float    v8f;

// ---------------------------------------------------------------- utilities
__global__ void k_zero_u32(unsigned* __restrict__ p, long n) {
    long i = (long)blockIdx.x * blockDim.x + threadIdx.x;
    if (i < n) p[i] = 0u;
}

__global__ void k_cvt_f16(const float* __restrict__ src, _Float16* __restrict__ dst, long n) {
    long i = (long)blockIdx.x * blockDim.x + threadIdx.x;
    if (i < n) dst[i] = (_Float16)src[i];
}

// W [K][Nc] f32 row-major -> WT [Nc][K] f16 (transposed, for contiguous B fragments)
__global__ void k_cvt_w_t(const float* __restrict__ W, _Float16* __restrict__ WT, int K, int Nc) {
    int i = blockIdx.x * blockDim.x + threadIdx.x;
    if (i >= K * Nc) return;
    int k = i / Nc, n = i - k * Nc;
    WT[(size_t)n * K + k] = (_Float16)W[i];
}

// ------------------------------------------------- dual WMMA GEMM (C0,C1 = A*B0, A*B1)
// A: [M][K] f16 row-major. B0T/B1T: [Nc][K] f16 (i.e. B transposed). C: [M][Nc] f32.
// One wave per 16x16 output tile; shared A fragment feeds two WMMA accumulators.
__global__ void __launch_bounds__(256)
k_wmma_dual_gemm(const _Float16* __restrict__ A,
                 const _Float16* __restrict__ B0T,
                 const _Float16* __restrict__ B1T,
                 float* __restrict__ C0, float* __restrict__ C1,
                 int M, int Nc, int K)
{
    const int lane  = threadIdx.x & 31;
    const int wid   = blockIdx.x * (blockDim.x >> 5) + (threadIdx.x >> 5);
    const int ntn   = Nc >> 4;
    const int mtile = wid / ntn;
    const int ntile = wid - mtile * ntn;
    if (mtile * 16 >= M) return;

    const int lan15 = lane & 15;
    const int khalf = lane >> 4;          // which half of the wave
    int mrow = mtile * 16 + lan15;
    if (mrow >= M) mrow = M - 1;          // safety clamp (M is a multiple of 16 here)
    const int ncol = ntile * 16 + lan15;

    const _Float16* arow = A   + (size_t)mrow * K;
    const _Float16* b0r  = B0T + (size_t)ncol * K + khalf * 16;
    const _Float16* b1r  = B1T + (size_t)ncol * K + khalf * 16;

    v8f acc0 = {}; v8f acc1 = {};
    for (int kb = 0; kb < K; kb += 32) {
        // A fragment: lanes 0-15 hold K kb+[0..15], lanes 16-31 hold kb+[16..31];
        // one contiguous 32B vector load (matches B's K order).
        v16h a  = *(const v16h*)(arow + kb + khalf * 16);
        // B fragment: lane half selects K group of 16; contiguous 32B per lane from BT.
        v16h b0 = *(const v16h*)(b0r + kb);
        v16h b1 = *(const v16h*)(b1r + kb);
        __builtin_prefetch(arow + kb + 64, 0, 0);   // global_prefetch_b8 for next k-step

        acc0 = __builtin_amdgcn_wmma_f32_16x16x32_f16(false, a, false, b0,
                                                      (short)0, acc0, false, false);
        acc1 = __builtin_amdgcn_wmma_f32_16x16x32_f16(false, a, false, b1,
                                                      (short)0, acc1, false, false);
    }
    // C/D layout: VGPR i -> row i (+8 for upper lane half), col = lane&15
    #pragma unroll
    for (int i = 0; i < 8; ++i) {
        int row = mtile * 16 + i + 8 * khalf;
        if (row < M) {
            C0[(size_t)row * Nc + ncol] = acc0[i];
            C1[(size_t)row * Nc + ncol] = acc1[i];
        }
    }
}

// ---------------------------------------------------------------- edge phase
__device__ __forceinline__ unsigned enc_ord(float f) {
    unsigned u = __float_as_uint(f);
    return (u & 0x80000000u) ? ~u : (u | 0x80000000u);
}
__device__ __forceinline__ float dec_ord(unsigned u) {
    unsigned bits = (u & 0x80000000u) ? (u & 0x7fffffffu) : ~u;
    return __uint_as_float(bits);
}

// score[e,h] = att[h,:] . leaky_relu(xl[src] + xr[dst]); also segment-max via atomicMax
__global__ void k_edge_score(const float* __restrict__ xl, const float* __restrict__ xr,
                             const float* __restrict__ att, const int* __restrict__ ei,
                             int E, int Etot, int H, int C,
                             float* __restrict__ sc, unsigned* __restrict__ smax)
{
    int tid = blockIdx.x * blockDim.x + threadIdx.x;
    if (tid >= Etot * H) return;
    int e = tid / H, h = tid - e * H;
    int s, d;
    if (e < E) { s = ei[e]; d = ei[E + e]; } else { s = d = e - E; }
    const int D = H * C;
    const float* pl = xl + (size_t)s * D + h * C;
    const float* pr = xr + (size_t)d * D + h * C;
    const float* pa = att + h * C;
    float acc = 0.f;
    for (int c = 0; c < C; ++c) {
        float v = pl[c] + pr[c];
        v = v > 0.f ? v : 0.2f * v;            // leaky_relu(0.2)
        acc += pa[c] * v;
    }
    sc[tid] = acc;
    atomicMax(&smax[(size_t)d * H + h], enc_ord(acc));
}

// ex = exp(score - smax[dst]); denom[dst] += ex   (sc overwritten with ex)
__global__ void k_edge_exp(float* __restrict__ sc, const unsigned* __restrict__ smax,
                           float* __restrict__ den, const int* __restrict__ ei,
                           int E, int Etot, int H)
{
    int tid = blockIdx.x * blockDim.x + threadIdx.x;
    if (tid >= Etot * H) return;
    int e = tid / H, h = tid - e * H;
    int d = (e < E) ? ei[E + e] : e - E;
    float m  = dec_ord(smax[(size_t)d * H + h]);
    float ex = __expf(sc[tid] - m);
    sc[tid] = ex;
    atomicAdd(&den[(size_t)d * H + h], ex);
}

// out[dst] += alpha * xl[src]
__global__ void k_edge_agg(const float* __restrict__ sc, const float* __restrict__ den,
                           const float* __restrict__ xl, const int* __restrict__ ei,
                           int E, int Etot, int H, int C, float* __restrict__ out)
{
    int tid = blockIdx.x * blockDim.x + threadIdx.x;
    if (tid >= Etot * H) return;
    int e = tid / H, h = tid - e * H;
    int s, d;
    if (e < E) { s = ei[e]; d = ei[E + e]; } else { s = d = e - E; }
    const int D = H * C;
    float alpha = sc[tid] / (den[(size_t)d * H + h] + 1e-16f);
    const float* pl = xl + (size_t)s * D + h * C;
    float* po = out + (size_t)d * D + h * C;
    for (int c = 0; c < C; ++c)
        atomicAdd(&po[c], alpha * pl[c]);
}

// h = elu(h + b); also emit f16 copy for the next WMMA layer
__global__ void k_bias_elu_f16(float* __restrict__ h, const float* __restrict__ b,
                               _Float16* __restrict__ hh, long n, int D)
{
    long i = (long)blockIdx.x * blockDim.x + threadIdx.x;
    if (i >= n) return;
    float v = h[i] + b[(int)(i % D)];
    v = v > 0.f ? v : (__expf(v) - 1.f);
    h[i]  = v;
    hh[i] = (_Float16)v;
}

__global__ void k_bias(float* __restrict__ h, const float* __restrict__ b, long n, int D)
{
    long i = (long)blockIdx.x * blockDim.x + threadIdx.x;
    if (i >= n) return;
    h[i] += b[(int)(i % D)];
}

// ---------------------------------------------------------------- norm + softmax
// one block per channel: mean / rstd over node dimension (population variance)
__global__ void k_inorm_stats(const float* __restrict__ h, int N, int D,
                              float* __restrict__ stats)
{
    __shared__ float ssum[256];
    __shared__ float ssq[256];
    const int c = blockIdx.x;
    float s = 0.f, q = 0.f;
    for (int n = threadIdx.x; n < N; n += blockDim.x) {
        float v = h[(size_t)n * D + c];
        s += v; q += v * v;
    }
    ssum[threadIdx.x] = s; ssq[threadIdx.x] = q;
    __syncthreads();
    for (int st = 128; st > 0; st >>= 1) {
        if ((int)threadIdx.x < st) {
            ssum[threadIdx.x] += ssum[threadIdx.x + st];
            ssq[threadIdx.x]  += ssq[threadIdx.x + st];
        }
        __syncthreads();
    }
    if (threadIdx.x == 0) {
        float mean = ssum[0] / (float)N;
        float var  = ssq[0] / (float)N - mean * mean;
        stats[c]     = mean;
        stats[D + c] = rsqrtf(var + 1e-5f);
    }
}

__global__ void k_inorm_apply(const float* __restrict__ h, const float* __restrict__ stats,
                              const float* __restrict__ gamma, const float* __restrict__ beta,
                              float* __restrict__ out, long n, int D)
{
    long i = (long)blockIdx.x * blockDim.x + threadIdx.x;
    if (i >= n) return;
    int c = (int)(i % D);
    out[i] = (h[i] - stats[c]) * stats[D + c] * gamma[c] + beta[c];
}

// one wave32 per node; 64 channels => 2 per lane; shuffle reductions
__global__ void k_log_softmax64(const float* __restrict__ hn, float* __restrict__ out, int N)
{
    const int lane = threadIdx.x & 31;
    const int node = blockIdx.x * (blockDim.x >> 5) + (threadIdx.x >> 5);
    if (node >= N) return;
    const float* row = hn + (size_t)node * 64;
    float v0 = row[lane], v1 = row[lane + 32];
    float m = fmaxf(v0, v1);
    #pragma unroll
    for (int o = 16; o > 0; o >>= 1) m = fmaxf(m, __shfl_xor(m, o, 32));
    float s = __expf(v0 - m) + __expf(v1 - m);
    #pragma unroll
    for (int o = 16; o > 0; o >>= 1) s += __shfl_xor(s, o, 32);
    float lse = m + __logf(s);
    out[(size_t)node * 64 + lane]      = v0 - lse;
    out[(size_t)node * 64 + lane + 32] = v1 - lse;
}

// ---------------------------------------------------------------- launcher
extern "C" void kernel_launch(void* const* d_in, const int* in_sizes, int n_in,
                              void* d_out, int out_size, void* d_ws, size_t ws_size,
                              hipStream_t stream)
{
    (void)n_in; (void)out_size; (void)ws_size;
    const float* x     = (const float*)d_in[0];
    const float* Wl1   = (const float*)d_in[1];
    const float* Wr1   = (const float*)d_in[2];
    const float* att1  = (const float*)d_in[3];
    const float* b1    = (const float*)d_in[4];
    const float* Wl2   = (const float*)d_in[5];
    const float* Wr2   = (const float*)d_in[6];
    const float* att2  = (const float*)d_in[7];
    const float* b2    = (const float*)d_in[8];
    const float* gamma = (const float*)d_in[9];
    const float* beta  = (const float*)d_in[10];
    const int*   ei    = (const int*)d_in[11];

    const int F  = 128, H1 = 8, C1 = 32, D1 = 256, D2 = 64;
    const int N    = in_sizes[0] / F;
    const int E    = in_sizes[11] / 2;
    const int Etot = E + N;                       // add_self_loops=True

    // ---- workspace arena (phase-2 scratch aliases phase-1 scratch) ----
    char* ws = (char*)d_ws;
    size_t off = 0;
    auto alloc = [&](size_t bytes) -> char* {
        char* p = ws + off;
        off = (off + bytes + 255) & ~(size_t)255;
        return p;
    };
    // persistent across phases
    float*    h1    = (float*)   alloc((size_t)N * D1 * 4);
    _Float16* h1h   = (_Float16*)alloc((size_t)N * D1 * 2);
    float*    stats = (float*)   alloc((size_t)2 * D2 * 4);
    const size_t mark = off;
    // phase 1
    _Float16* xh    = (_Float16*)alloc((size_t)N * F * 2);
    _Float16* wl1t  = (_Float16*)alloc((size_t)D1 * F * 2);
    _Float16* wr1t  = (_Float16*)alloc((size_t)D1 * F * 2);
    float*    xl1   = (float*)   alloc((size_t)N * D1 * 4);
    float*    xr1   = (float*)   alloc((size_t)N * D1 * 4);
    float*    sc1   = (float*)   alloc((size_t)Etot * H1 * 4);
    unsigned* smax1 = (unsigned*)alloc((size_t)N * H1 * 4);
    float*    den1  = (float*)   alloc((size_t)N * H1 * 4);
    // phase 2 (aliases phase 1 scratch; used strictly after it)
    off = mark;
    _Float16* wl2t  = (_Float16*)alloc((size_t)D2 * D1 * 2);
    _Float16* wr2t  = (_Float16*)alloc((size_t)D2 * D1 * 2);
    float*    xl2   = (float*)   alloc((size_t)N * D2 * 4);
    float*    xr2   = (float*)   alloc((size_t)N * D2 * 4);
    float*    sc2   = (float*)   alloc((size_t)Etot * 4);
    unsigned* smax2 = (unsigned*)alloc((size_t)N * 4);
    float*    den2  = (float*)   alloc((size_t)N * 4);
    float*    h2    = (float*)   alloc((size_t)N * D2 * 4);

    auto cdiv = [](long a, long b) { return (int)((a + b - 1) / b); };
    const int T = 256;

    // ---------------- layer 1 ----------------
    k_zero_u32<<<cdiv((long)N * D1, T), T, 0, stream>>>((unsigned*)h1, (long)N * D1);
    k_zero_u32<<<cdiv((long)N * H1, T), T, 0, stream>>>(smax1, (long)N * H1);
    k_zero_u32<<<cdiv((long)N * H1, T), T, 0, stream>>>((unsigned*)den1, (long)N * H1);

    k_cvt_f16 <<<cdiv((long)N * F, T), T, 0, stream>>>(x, xh, (long)N * F);
    k_cvt_w_t <<<cdiv((long)F * D1, T), T, 0, stream>>>(Wl1, wl1t, F, D1);
    k_cvt_w_t <<<cdiv((long)F * D1, T), T, 0, stream>>>(Wr1, wr1t, F, D1);

    {   // xl1 = x@Wl1, xr1 = x@Wr1 (WMMA, shared A fragments)
        int tiles = ((N + 15) / 16) * (D1 / 16);
        k_wmma_dual_gemm<<<cdiv(tiles, 8), 256, 0, stream>>>(xh, wl1t, wr1t,
                                                             xl1, xr1, N, D1, F);
    }

    k_edge_score<<<cdiv((long)Etot * H1, T), T, 0, stream>>>(xl1, xr1, att1, ei,
                                                             E, Etot, H1, C1, sc1, smax1);
    k_edge_exp  <<<cdiv((long)Etot * H1, T), T, 0, stream>>>(sc1, smax1, den1, ei,
                                                             E, Etot, H1);
    k_edge_agg  <<<cdiv((long)Etot * H1, T), T, 0, stream>>>(sc1, den1, xl1, ei,
                                                             E, Etot, H1, C1, h1);
    k_bias_elu_f16<<<cdiv((long)N * D1, T), T, 0, stream>>>(h1, b1, h1h, (long)N * D1, D1);

    // ---------------- layer 2 ----------------
    k_cvt_w_t<<<cdiv((long)D1 * D2, T), T, 0, stream>>>(Wl2, wl2t, D1, D2);
    k_cvt_w_t<<<cdiv((long)D1 * D2, T), T, 0, stream>>>(Wr2, wr2t, D1, D2);

    {   // xl2 = h1@Wl2, xr2 = h1@Wr2
        int tiles = ((N + 15) / 16) * (D2 / 16);
        k_wmma_dual_gemm<<<cdiv(tiles, 8), 256, 0, stream>>>(h1h, wl2t, wr2t,
                                                             xl2, xr2, N, D2, D1);
    }

    k_zero_u32<<<cdiv((long)N, T), T, 0, stream>>>(smax2, (long)N);
    k_zero_u32<<<cdiv((long)N, T), T, 0, stream>>>((unsigned*)den2, (long)N);
    k_zero_u32<<<cdiv((long)N * D2, T), T, 0, stream>>>((unsigned*)h2, (long)N * D2);

    k_edge_score<<<cdiv((long)Etot, T), T, 0, stream>>>(xl2, xr2, att2, ei,
                                                        E, Etot, 1, D2, sc2, smax2);
    k_edge_exp  <<<cdiv((long)Etot, T), T, 0, stream>>>(sc2, smax2, den2, ei, E, Etot, 1);
    k_edge_agg  <<<cdiv((long)Etot, T), T, 0, stream>>>(sc2, den2, xl2, ei,
                                                        E, Etot, 1, D2, h2);
    k_bias<<<cdiv((long)N * D2, T), T, 0, stream>>>(h2, b2, (long)N * D2, D2);

    // ---------------- instance norm + log_softmax ----------------
    float* out_h  = (float*)d_out;
    float* out_ls = out_h + (size_t)N * D2;
    k_inorm_stats<<<D2, 256, 0, stream>>>(h2, N, D2, stats);
    k_inorm_apply<<<cdiv((long)N * D2, T), T, 0, stream>>>(h2, stats, gamma, beta,
                                                           out_h, (long)N * D2, D2);
    k_log_softmax64<<<cdiv((long)N, 8), 256, 0, stream>>>(out_h, out_ls, N);
}
